// SingleAttention_2456721293785
// MI455X (gfx1250) — matrix operand order repaired and checked by reading, not compiled, not run on hardware
//
#include <hip/hip_runtime.h>
#include <hip/hip_bf16.h>
#include <stdint.h>

typedef __bf16 bf16_t;
typedef __attribute__((ext_vector_type(16))) __bf16 v16bf;
typedef __attribute__((ext_vector_type(8)))  float  v8f;

#define D_MODEL 256
#define SEQ     4096
#define BATCH   4
#define SBLK    32     // keys per pipeline step
#define DVB     128    // dv columns per block (grid.y splits 256 into 2)

union ARegs { v16bf v; uint32_t u[8]; };

__device__ inline uint32_t pack2_bf16(float lo, float hi) {
  union { uint32_t u; __bf16 h[2]; } r;
  r.h[0] = (__bf16)lo; r.h[1] = (__bf16)hi;
  return r.u;
}

// A operand (16x32 bf16) from fp32 row-major source.
// Lane L: M = L&15, khalf = L>>4. VGPR v holds K = (v/4)*16 + khalf*8 + (v%4)*2 (+1).
__device__ inline v16bf load_A_f32(const float* __restrict__ base, int ld, int lane) {
  int M = lane & 15, kh = lane >> 4;
  ARegs r;
  const float* p = base + (size_t)M * ld;
#pragma unroll
  for (int v = 0; v < 8; ++v) {
    int kb = ((v >> 2) << 4) + (kh << 3) + ((v & 3) << 1);
    r.u[v] = pack2_bf16(p[kb], p[kb + 1]);
  }
  return r.v;
}

// A operand (16x32) from bf16 row-major source (K-pairs contiguous -> b32 loads).
__device__ inline v16bf load_A_bf16(const bf16_t* base, int ld, int lane) {
  int M = lane & 15, kh = lane >> 4;
  ARegs r;
#pragma unroll
  for (int v = 0; v < 8; ++v) {
    int kb = ((v >> 2) << 4) + (kh << 3) + ((v & 3) << 1);
    r.u[v] = *(const uint32_t*)(base + (size_t)M * ld + kb);
  }
  return r.v;
}

// B operand (32x16 bf16) from B^T stored N-major (K contiguous per row):
// element (K,N) at baseT[N*ld + K]. Lane L: N = L&15, khalf = L>>4; VGPR v: K = (v/4)*16 + khalf*8 + (v%4)*2.
__device__ inline v16bf load_B_bf16(const bf16_t* baseT, int ld, int lane) {
  int N = lane & 15, kh = lane >> 4;
  ARegs r;
  const bf16_t* p = baseT + (size_t)N * ld + kh * 8;
#pragma unroll
  for (int v = 0; v < 8; ++v) r.u[v] = *(const uint32_t*)(p + ((v >> 2) << 4) + ((v & 3) << 1));
  return r.v;
}

// ---- CDNA5 async copy helpers ---------------------------------------------
// Low 32 bits of a shared-aperture generic pointer are the LDS byte offset
// (clang lowers AS3->generic addrspacecast as {aperture_hi, lds_offset}).
__device__ inline uint32_t lds_off(const void* p) {
  return (uint32_t)(uintptr_t)p;
}
__device__ inline void async_load_b128(uint32_t lds, const void* gaddr) {
  asm volatile("global_load_async_to_lds_b128 %0, %1, off"
               :: "v"(lds), "v"(gaddr) : "memory");
}
template <int N> __device__ inline void wait_async() {
#if defined(__has_builtin) && __has_builtin(__builtin_amdgcn_s_wait_asynccnt)
  __builtin_amdgcn_s_wait_asynccnt(N);
#else
  asm volatile("s_wait_asynccnt %0" :: "n"(N) : "memory");
#endif
}

// ---- Kernel 1: convert + transpose weights: Wt[k][d] = bf16(W[d][k]) -------
__global__ void prep_weights(const float* __restrict__ Wq, const float* __restrict__ Wk,
                             const float* __restrict__ Wv, bf16_t* __restrict__ WtAll) {
  int w = blockIdx.y;
  const float* W = (w == 0) ? Wq : (w == 1) ? Wk : Wv;
  bf16_t* Wt = WtAll + (size_t)w * D_MODEL * D_MODEL;
  int idx = blockIdx.x * blockDim.x + threadIdx.x;   // 0..65535
  int k = idx >> 8, d = idx & 255;
  Wt[idx] = (bf16_t)W[d * D_MODEL + k];
}

// ---- Kernel 2: projection  out = (X @ W + b) * scale  (bf16 out) -----------
template <int TRANSPOSED>
__global__ void __launch_bounds__(128)
proj_kernel(const float* __restrict__ X, const bf16_t* __restrict__ Wt,
            const float* __restrict__ bias, bf16_t* __restrict__ out, float scale) {
  int lane = threadIdx.x & 31;
  int wid  = threadIdx.x >> 5;
  int r0   = (blockIdx.x * 4 + wid) * 16;
  int kh   = lane >> 4;
  const float* A = X + (size_t)r0 * D_MODEL;

  v16bf a[8];
#pragma unroll
  for (int ks = 0; ks < 8; ++ks) a[ks] = load_A_f32(A + ks * 32, D_MODEL, lane);

#pragma unroll 1
  for (int nt = 0; nt < 16; ++nt) {
    v8f acc = {};
    const bf16_t* Bt = Wt + (size_t)nt * 16 * D_MODEL;
#pragma unroll
    for (int ks = 0; ks < 8; ++ks) {
      v16bf b = load_B_bf16(Bt + ks * 32, D_MODEL, lane);
      acc = __builtin_amdgcn_wmma_f32_16x16x32_bf16(false, a[ks], false, b,
                                                    (short)0, acc, false, false);
    }
    int col = nt * 16 + (lane & 15);
    float bv = bias[col];
#pragma unroll
    for (int i = 0; i < 8; ++i) {
      float val = (acc[i] + bv) * scale;
      int row = r0 + i + (kh << 3);
      if (!TRANSPOSED) {
        out[(size_t)row * D_MODEL + col] = (bf16_t)val;             // [row][256]
      } else {
        int b_ = row >> 12;                                          // row / SEQ
        int s  = row & (SEQ - 1);
        out[((size_t)b_ * D_MODEL + col) * SEQ + s] = (bf16_t)val;   // Vt[b][dv][s]
      }
    }
  }
}

// ---- Kernel 3: flash attention with async-LDS double-buffered K/V ---------
// Block: 8 waves = 8 query tiles (same batch); K/V tiles staged in LDS via
// global_load_async_to_lds_b128, pipelined one step ahead (ASYNCcnt).
__global__ void __launch_bounds__(256)
attn_kernel(const bf16_t* __restrict__ Q, const bf16_t* __restrict__ K,
            const bf16_t* __restrict__ Vt, float* __restrict__ out) {
  __shared__ __align__(16) bf16_t kbuf[2][SBLK][D_MODEL];  // 2 x 16 KB
  __shared__ __align__(16) bf16_t vbuf[2][DVB][SBLK];      // 2 x  8 KB
  __shared__ __align__(16) bf16_t pbuf[8][16 * SBLK];      //      8 KB

  int tid  = threadIdx.x;
  int lane = tid & 31;
  int wid  = tid >> 5;
  int tile = blockIdx.x * 8 + wid;            // query tile (256 per batch)
  int b    = tile >> 8;
  int q0g  = tile * 16;                       // global query row
  int dv0  = blockIdx.y * DVB;
  int kh   = lane >> 4;

  const bf16_t* Kb  = K  + (size_t)b * SEQ * D_MODEL;
  const bf16_t* Vtb = Vt + (size_t)b * D_MODEL * SEQ;

  // issue one K/V tile into LDS buffer `buf` (6 async b128 ops per wave)
  auto issue_tile = [&](int s0, int buf) {
    const char* kg = (const char*)(Kb + (size_t)s0 * D_MODEL);   // contiguous 16 KB
    uint32_t kl = lds_off(&kbuf[buf][0][0]);
#pragma unroll
    for (int j = 0; j < 4; ++j) {
      int off = tid * 16 + j * 4096;
      async_load_b128(kl + off, kg + off);
    }
    uint32_t vl = lds_off(&vbuf[buf][0][0]);
#pragma unroll
    for (int j = 0; j < 2; ++j) {
      int c = tid + j * 256;                   // 512 chunks of 16 B
      int row = c >> 2, co = (c & 3) * 16;     // 4 chunks per 64 B row
      async_load_b128(vl + row * (SBLK * 2) + co,
                      (const char*)(Vtb + (size_t)(dv0 + row) * SEQ + s0) + co);
    }
  };

  v16bf qa[8];
#pragma unroll
  for (int ks = 0; ks < 8; ++ks)
    qa[ks] = load_A_bf16(Q + (size_t)q0g * D_MODEL + ks * 32, D_MODEL, lane);

  v8f o[8];
  float m[8], l[8];
#pragma unroll
  for (int t = 0; t < 8; ++t)
#pragma unroll
    for (int i = 0; i < 8; ++i) o[t][i] = 0.f;
#pragma unroll
  for (int i = 0; i < 8; ++i) { m[i] = -1e30f; l[i] = 0.f; }

  bf16_t* pw = pbuf[wid];
  issue_tile(0, 0);

#pragma unroll 1
  for (int it = 0; it < SEQ / SBLK; ++it) {
    int cur = it & 1;
    if (it + 1 < SEQ / SBLK) {
      issue_tile((it + 1) * SBLK, cur ^ 1);
      wait_async<6>();                         // previous tile's 6 ops landed
    } else {
      wait_async<0>();
    }
    __syncthreads();                           // all waves' DMA portions visible

    const bf16_t* kc = &kbuf[cur][0][0];
    const bf16_t* vc = &vbuf[cur][0][0];

    // scores: two 16x16 f32 tiles over 32 keys (1/sqrt(dk) folded into Q)
    v8f c0 = {}, c1 = {};
#pragma unroll
    for (int ks = 0; ks < 8; ++ks) {
      v16bf b0 = load_B_bf16(kc + ks * 32, D_MODEL, lane);
      c0 = __builtin_amdgcn_wmma_f32_16x16x32_bf16(false, qa[ks], false, b0,
                                                   (short)0, c0, false, false);
    }
#pragma unroll
    for (int ks = 0; ks < 8; ++ks) {
      v16bf b1 = load_B_bf16(kc + 16 * D_MODEL + ks * 32, D_MODEL, lane);
      c1 = __builtin_amdgcn_wmma_f32_16x16x32_bf16(false, qa[ks], false, b1,
                                                   (short)0, c1, false, false);
    }
    // online softmax (row reductions stay inside each 16-lane half)
    float alpha[8];
#pragma unroll
    for (int i = 0; i < 8; ++i) {
      float mx = fmaxf(c0[i], c1[i]);
      mx = fmaxf(mx, __shfl_xor(mx, 1));
      mx = fmaxf(mx, __shfl_xor(mx, 2));
      mx = fmaxf(mx, __shfl_xor(mx, 4));
      mx = fmaxf(mx, __shfl_xor(mx, 8));
      float mn = fmaxf(m[i], mx);
      alpha[i] = __expf(m[i] - mn);
      m[i] = mn;
      float p0 = __expf(c0[i] - mn);
      float p1 = __expf(c1[i] - mn);
      c0[i] = p0; c1[i] = p1;
      float rs = p0 + p1;
      rs += __shfl_xor(rs, 1);
      rs += __shfl_xor(rs, 2);
      rs += __shfl_xor(rs, 4);
      rs += __shfl_xor(rs, 8);
      l[i] = l[i] * alpha[i] + rs;
#pragma unroll
      for (int t = 0; t < 8; ++t) o[t][i] *= alpha[i];
    }
    // P: C-layout -> row-major bf16 in LDS -> reload in A-layout
    int N = lane & 15;
#pragma unroll
    for (int i = 0; i < 8; ++i) {
      int row = i + (kh << 3);
      pw[row * SBLK + N]      = (bf16_t)c0[i];
      pw[row * SBLK + 16 + N] = (bf16_t)c1[i];
    }
    __syncthreads();
    v16bf pa = load_A_bf16(pw, SBLK, lane);
    __syncthreads();
    // O += P @ V   (vbuf is dv-major, K contiguous)
#pragma unroll
    for (int t = 0; t < 8; ++t) {
      v16bf bvv = load_B_bf16(vc + (size_t)(t * 16) * SBLK, SBLK, lane);
      o[t] = __builtin_amdgcn_wmma_f32_16x16x32_bf16(false, pa, false, bvv,
                                                     (short)0, o[t], false, false);
    }
    __syncthreads();   // protect buffer cur reuse by next iteration's issue
  }

  // epilogue: divide by row sums, store f32
#pragma unroll
  for (int t = 0; t < 8; ++t) {
    int col = dv0 + t * 16 + (lane & 15);
#pragma unroll
    for (int i = 0; i < 8; ++i) {
      int row = q0g + i + (kh << 3);
      out[(size_t)row * D_MODEL + col] = o[t][i] / l[i];
    }
  }
}

extern "C" void kernel_launch(void* const* d_in, const int* in_sizes, int n_in,
                              void* d_out, int out_size, void* d_ws, size_t ws_size,
                              hipStream_t stream) {
  const float* q_in = (const float*)d_in[0];
  const float* k_in = (const float*)d_in[1];
  const float* v_in = (const float*)d_in[2];
  const float* Wq   = (const float*)d_in[3];
  const float* bq   = (const float*)d_in[4];
  const float* Wk   = (const float*)d_in[5];
  const float* bk   = (const float*)d_in[6];
  const float* Wv   = (const float*)d_in[7];
  const float* bv   = (const float*)d_in[8];
  float* out = (float*)d_out;

  char* ws = (char*)d_ws;
  bf16_t* Wt  = (bf16_t*)(ws);                            // 3 * 64K bf16 = 384 KB
  bf16_t* Qb  = (bf16_t*)(ws + 393216);                   // 8 MB
  bf16_t* Kb  = (bf16_t*)(ws + 393216 + 8388608);         // 8 MB
  bf16_t* Vtb = (bf16_t*)(ws + 393216 + 2 * 8388608);     // 8 MB (transposed)

  prep_weights<<<dim3(256, 3), 256, 0, stream>>>(Wq, Wk, Wv, Wt);
  proj_kernel<0><<<dim3(256), 128, 0, stream>>>(q_in, Wt,          bq, Qb,  0.0625f);
  proj_kernel<0><<<dim3(256), 128, 0, stream>>>(k_in, Wt + 65536,  bk, Kb,  1.0f);
  proj_kernel<1><<<dim3(256), 128, 0, stream>>>(v_in, Wt + 131072, bv, Vtb, 1.0f);
  attn_kernel<<<dim3(128, 2), 256, 0, stream>>>(Qb, Kb, Vtb, out);
}
